// sLSTMCell_979252544372
// MI455X (gfx1250) — matrix operand order repaired and checked by reading, not compiled, not run on hardware
//
#include <hip/hip_runtime.h>
#include <hip/hip_bf16.h>

// ---------------------------------------------------------------------------
// sLSTM cell, B=8192, I=H=1024, for MI455X (gfx1250, wave32, WMMA)
//   pregate_g = x @ W_g^T + h @ R_g^T + b_g   (4 gates fused per workgroup)
//   + stabilized sLSTM pointwise epilogue fused into the GEMM.
// bf16 inputs / f32 accumulation via v_wmma_f32_16x16x32_bf16.
// Tile staging uses GLOBAL_LOAD_ASYNC_TO_LDS_B128 (ASYNCcnt) when available.
// ---------------------------------------------------------------------------

#define B_DIM   8192
#define K_CAT   2048      // I + H concatenated reduction dim
#define N_GATE  1024      // per-gate output width (H)
#define EPSF    1e-7f

typedef __attribute__((ext_vector_type(16))) __bf16 v16bf;
typedef __attribute__((ext_vector_type(8)))  float  v8f;
typedef __attribute__((ext_vector_type(4)))  int    v4i_t;

union Frag32B {           // 32 bytes = 16 bf16 = one WMMA A/B fragment
    uint4  q[2];
    v16bf  v;
};

#if defined(__HIP_DEVICE_COMPILE__) && defined(__gfx1250__) && \
    __has_builtin(__builtin_amdgcn_global_load_async_to_lds_b128)
#define HAS_ASYNC_LDS 1
#else
#define HAS_ASYNC_LDS 0
#endif

// one-instruction 16B global -> LDS copy (per lane), ASYNCcnt-tracked
__device__ __forceinline__ void async_cp16(void* lds_dst, const void* gsrc) {
#if HAS_ASYNC_LDS
    __builtin_amdgcn_global_load_async_to_lds_b128(
        (__attribute__((address_space(1))) v4i_t*)gsrc,
        (__attribute__((address_space(3))) v4i_t*)lds_dst,
        0, 0);
#else
    *(uint4*)lds_dst = *(const uint4*)gsrc;
#endif
}

__device__ __forceinline__ void wait_async0() {
#if HAS_ASYNC_LDS
#if __has_builtin(__builtin_amdgcn_s_wait_asynccnt)
    __builtin_amdgcn_s_wait_asynccnt(0);
#else
    asm volatile("s_wait_asynccnt 0x0" ::: "memory");
#endif
#endif
}

__device__ __forceinline__ unsigned short f32_to_bf16_rne(float f) {
    unsigned int u = __float_as_uint(f);
    unsigned int r = u + 0x7FFFu + ((u >> 16) & 1u);
    return (unsigned short)(r >> 16);
}
__device__ __forceinline__ unsigned int pack2_bf16(float a, float b) {
    return (unsigned int)f32_to_bf16_rne(a) |
           ((unsigned int)f32_to_bf16_rne(b) << 16);
}

// ---- pass 1: pack [x | h_prev] -> bf16 A-buffer (8192 x 2048) --------------
__global__ void pack_xh(const float* __restrict__ x,
                        const float* __restrict__ h,
                        unsigned short* __restrict__ Abuf) {
    size_t tid = (size_t)blockIdx.x * blockDim.x + threadIdx.x;
    size_t e   = tid * 8;                       // 8 elems / thread
    size_t row = e >> 11;                       // / 2048
    size_t col = e & 2047;
    const float* src = (col < 1024) ? (x + row * 1024 + col)
                                    : (h + row * 1024 + (col - 1024));
    float4 f0 = ((const float4*)src)[0];
    float4 f1 = ((const float4*)src)[1];
    uint4 o;
    o.x = pack2_bf16(f0.x, f0.y);
    o.y = pack2_bf16(f0.z, f0.w);
    o.z = pack2_bf16(f1.x, f1.y);
    o.w = pack2_bf16(f1.z, f1.w);
    *(uint4*)(Abuf + e) = o;
}

// ---- pass 2: pack [W_g | R_g] per gate -> bf16 B-buffer [4][1024][2048] ----
__global__ void pack_wr(const float* __restrict__ w0, const float* __restrict__ r0,
                        const float* __restrict__ w1, const float* __restrict__ r1,
                        const float* __restrict__ w2, const float* __restrict__ r2,
                        const float* __restrict__ w3, const float* __restrict__ r3,
                        unsigned short* __restrict__ Bbuf) {
    size_t tid = (size_t)blockIdx.x * blockDim.x + threadIdx.x;
    size_t e   = tid * 8;                       // into [4][1024][2048]
    int    g   = (int)(e >> 21);                // / (1024*2048)
    size_t rem = e & ((1ull << 21) - 1);
    size_t n   = rem >> 11;
    size_t col = rem & 2047;
    const float* wsel = (g == 0) ? w0 : (g == 1) ? w1 : (g == 2) ? w2 : w3;
    const float* rsel = (g == 0) ? r0 : (g == 1) ? r1 : (g == 2) ? r2 : r3;
    const float* src  = (col < 1024) ? (wsel + n * 1024 + col)
                                     : (rsel + n * 1024 + (col - 1024));
    float4 f0 = ((const float4*)src)[0];
    float4 f1 = ((const float4*)src)[1];
    uint4 o;
    o.x = pack2_bf16(f0.x, f0.y);
    o.y = pack2_bf16(f0.z, f0.w);
    o.z = pack2_bf16(f1.x, f1.y);
    o.w = pack2_bf16(f1.z, f1.w);
    *(uint4*)(Bbuf + e) = o;
}

// ---- pass 3: fused 4-gate GEMM + sLSTM pointwise ---------------------------
// grid: (N_GATE/32, B_DIM/64), block: 256 (8 waves, 4x2 wave grid)
// Each wave owns one 16x16 (M,N) position and 4 accumulators (one per gate).
__global__ __launch_bounds__(256)
void slstm_fused(const unsigned short* __restrict__ Abuf,
                 const unsigned short* __restrict__ Bbuf,
                 const float* __restrict__ bi, const float* __restrict__ bf_,
                 const float* __restrict__ bo, const float* __restrict__ bz,
                 const float* __restrict__ c_prev,
                 const float* __restrict__ n_prev,
                 const float* __restrict__ m_prev,
                 float* __restrict__ out) {
    // LDS tiles, rows padded 32->40 bf16 (80B) to spread banks; 16B aligned.
    __shared__ __align__(16) unsigned short As[2][64][40];
    __shared__ __align__(16) unsigned short Bs[2][4][32][40];

    const int t    = threadIdx.x;
    const int lane = t & 31;
    const int wid  = t >> 5;
    const int m0   = blockIdx.y * 64;
    const int n0   = blockIdx.x * 32;
    const int m_woff = (wid >> 1) * 16;
    const int n_woff = (wid & 1) * 16;

    // global->LDS staging assignments (one 16B A chunk + two 16B B chunks)
    const int lmA = t >> 2;              // A: row 0..63
    const int lkA = (t & 3) * 8;         // A: k-chunk of 8 bf16
    const int c0 = t, c1 = t + 256;      // B: two 8-elem chunks / thread
    const int g0 = c0 >> 7, r0i = (c0 >> 2) & 31, k0c = (c0 & 3) * 8;
    const int g1 = c1 >> 7, r1i = (c1 >> 2) & 31, k1c = (c1 & 3) * 8;

    const unsigned short* Ap  = Abuf + (size_t)(m0 + lmA) * K_CAT + lkA;
    const unsigned short* Bp0 = Bbuf + (size_t)g0 * (N_GATE * (size_t)K_CAT)
                                     + (size_t)(n0 + r0i) * K_CAT + k0c;
    const unsigned short* Bp1 = Bbuf + (size_t)g1 * (N_GATE * (size_t)K_CAT)
                                     + (size_t)(n0 + r1i) * K_CAT + k1c;

    // fragment addressing (ISA 7.12.2, 16-bit A 16x32 / B 32x16 layouts)
    const int mA = m_woff + (lane & 15);
    const int nB = n_woff + (lane & 15);
    const int kh = lane >> 4;

    v8f acc0 = {}, acc1 = {}, acc2 = {}, acc3 = {};

    // prime double buffer with k-step 0 (async DMA straight into LDS)
    async_cp16(&As[0][lmA][lkA],     Ap);   Ap  += 32;
    async_cp16(&Bs[0][g0][r0i][k0c], Bp0);  Bp0 += 32;
    async_cp16(&Bs[0][g1][r1i][k1c], Bp1);  Bp1 += 32;
    wait_async0();
    __syncthreads();

    for (int kt = 0; kt < K_CAT / 32; ++kt) {
        const int cur = kt & 1;
        if (kt < K_CAT / 32 - 1) {           // async-prefetch next k-step
            const int nxt = cur ^ 1;
            async_cp16(&As[nxt][lmA][lkA],     Ap);   Ap  += 32;
            async_cp16(&Bs[nxt][g0][r0i][k0c], Bp0);  Bp0 += 32;
            async_cp16(&Bs[nxt][g1][r1i][k1c], Bp1);  Bp1 += 32;
        }

        Frag32B fa;
        fa.q[0] = *(const uint4*)&As[cur][mA][kh * 8];
        fa.q[1] = *(const uint4*)&As[cur][mA][16 + kh * 8];

#define GATE_STEP(GI, ACC)                                                     \
        do {                                                                   \
            Frag32B fb;                                                        \
            fb.q[0] = *(const uint4*)&Bs[cur][GI][nB][kh * 8];                 \
            fb.q[1] = *(const uint4*)&Bs[cur][GI][nB][16 + kh * 8];            \
            ACC = __builtin_amdgcn_wmma_f32_16x16x32_bf16(                     \
                false, fa.v, false, fb.v, (short)0, ACC, false, false);        \
        } while (0)

        GATE_STEP(0, acc0);
        GATE_STEP(1, acc1);
        GATE_STEP(2, acc2);
        GATE_STEP(3, acc3);
#undef GATE_STEP

        wait_async0();                       // DMA for nxt complete
        __syncthreads();                     // all waves done reading cur
    }

    // ---- fused sLSTM epilogue; D layout: lane%16 -> N, (lane/16)*8+r -> M --
    const size_t BH = (size_t)B_DIM * N_GATE;
    const int ng = n0 + n_woff + (lane & 15);
    const float Bi = bi[ng], Bf = bf_[ng], Bo = bo[ng], Bz = bz[ng];

#pragma unroll
    for (int r = 0; r < 8; ++r) {
        const int    mg  = m0 + m_woff + (lane >> 4) * 8 + r;
        const size_t idx = (size_t)mg * N_GATE + ng;

        float it_ = fminf(fmaxf(acc0[r] + Bi, -50.f), 50.f);
        float ft_ = fminf(fmaxf(acc1[r] + Bf, -50.f), 50.f);
        float ot_ = fminf(fmaxf(acc2[r] + Bo, -50.f), 50.f);
        float zt_ = fminf(fmaxf(acc3[r] + Bz, -50.f), 50.f);

        float i_t  = __expf(it_);
        float fsig = 1.f / (1.f + __expf(-ft_));
        fsig = fminf(fmaxf(fsig, EPSF), 1.f - EPSF);
        float logf_ = __logf(fsig);

        float mp  = m_prev[idx];
        float m_t = fmaxf(logf_ + mp, __logf(i_t + EPSF));
        float i_pr = __expf(it_ - m_t);           // log(exp(it_)) == it_
        float f_pr = __expf(logf_ + mp - m_t);

        float c_t   = f_pr * c_prev[idx] + i_pr * tanhf(zt_);
        float n_t   = f_pr * n_prev[idx] + i_pr;
        float c_hat = c_t / (n_t + EPSF);
        float h_t   = (1.f / (1.f + __expf(-ot_))) * tanhf(c_hat);

        out[idx]          = h_t;   // h_t (primary output)
        out[BH + idx]     = h_t;   // h_t (state tuple)
        out[2 * BH + idx] = c_t;
        out[3 * BH + idx] = n_t;
        out[4 * BH + idx] = m_t;
    }
}

// ---------------------------------------------------------------------------
extern "C" void kernel_launch(void* const* d_in, const int* in_sizes, int n_in,
                              void* d_out, int out_size, void* d_ws, size_t ws_size,
                              hipStream_t stream) {
    // setup_inputs() dict order:
    // 0:x 1:h_prev 2:c_prev 3:n_prev 4:m_prev then per gate g in i,f,o,z:
    // w_g, r_g, b_g  -> indices 5+3g, 6+3g, 7+3g
    const float* x      = (const float*)d_in[0];
    const float* h_prev = (const float*)d_in[1];
    const float* c_prev = (const float*)d_in[2];
    const float* n_prev = (const float*)d_in[3];
    const float* m_prev = (const float*)d_in[4];
    const float* w[4]; const float* r[4]; const float* b[4];
    for (int g = 0; g < 4; ++g) {
        w[g] = (const float*)d_in[5 + 3 * g];
        r[g] = (const float*)d_in[6 + 3 * g];
        b[g] = (const float*)d_in[7 + 3 * g];
    }

    unsigned short* Abuf = (unsigned short*)d_ws;                 // 8192*2048 bf16
    unsigned short* Bbuf = Abuf + (size_t)B_DIM * K_CAT;          // 4*1024*2048 bf16

    {   // pack [x|h] -> bf16
        size_t elems = (size_t)B_DIM * K_CAT;
        dim3 grid((unsigned)(elems / 8 / 256));
        pack_xh<<<grid, 256, 0, stream>>>(x, h_prev, Abuf);
    }
    {   // pack [W|R] per gate -> bf16
        size_t elems = 4ull * N_GATE * K_CAT;
        dim3 grid((unsigned)(elems / 8 / 256));
        pack_wr<<<grid, 256, 0, stream>>>(w[0], r[0], w[1], r[1],
                                          w[2], r[2], w[3], r[3], Bbuf);
    }
    {   // fused GEMM + sLSTM
        dim3 grid(N_GATE / 32, B_DIM / 64);
        slstm_fused<<<grid, 256, 0, stream>>>(Abuf, Bbuf,
                                              b[0], b[1], b[2], b[3],
                                              c_prev, n_prev, m_prev,
                                              (float*)d_out);
    }
}
